// DynamicFilter_40312563040277
// MI455X (gfx1250) — matrix operand order repaired and checked by reading, not compiled, not run on hardware
//
#include <hip/hip_runtime.h>

#define DEVI __device__ __forceinline__

typedef __attribute__((ext_vector_type(16))) __bf16 v16bf;
typedef __attribute__((ext_vector_type(8)))  __bf16 v8bf;
typedef __attribute__((ext_vector_type(8)))  float  v8f;

// Problem sizes (fixed by the reference)
constexpr int Nn  = 32768;     // points per batch (N)
constexpr int BN  = 4 * Nn;    // total points
constexpr int NSHIFT = 15;     // log2(N)

// ---- LDS layout (bytes). All tensor bases 64B aligned; B-frag rows 32B aligned. ----
constexpr int O_W1T  = 0;                     // [32 n][96 k] bf16 (W1^T, k-padded 68->96)
constexpr int O_W2T  = O_W1T  + 32*96*2;      // [80 n][32 k] bf16 (W2^T, n-padded 68->80)
constexpr int O_WM1T = O_W2T  + 80*32*2;      // [64 n][96 k] bf16
constexpr int O_WM2T = O_WM1T + 64*96*2;      // [64 n][64 k] bf16
constexpr int O_WST  = O_WM2T + 64*64*2;      // [64 n][64 k] bf16
constexpr int O_GP   = O_WST  + 64*64*2;      // [16 wave][16 k][96 c] bf16 (gp, padded)
constexpr int O_FEAT = O_GP   + 16*16*96*2;   // [16 pt][96 c] bf16
constexpr int O_PTS  = O_FEAT + 16*96*2;      // [16 pt][64 c] bf16
constexpr int O_HB   = O_PTS  + 16*64*2;      // [16 wave][16 m][32 n] bf16 (hidden of GEMM1)
constexpr int O_HB2  = O_HB   + 16*16*32*2;   // [16 pt][64 n] bf16 (hidden of stage-2)
constexpr int O_MSK  = O_HB2  + 16*64*2;      // [16 wave][16 k] f32
constexpr int O_B1   = O_MSK  + 16*16*4;      // [32] f32
constexpr int O_B2   = O_B1   + 32*4;         // [80] f32 (padded)
constexpr int O_BM1  = O_B2   + 80*4;         // [64] f32
constexpr int O_BM2  = O_BM1  + 64*4;         // [64] f32
constexpr int O_BS   = O_BM2  + 64*4;         // [64] f32
constexpr int SMEM_BYTES = O_BS + 64*4;       // ~112.2 KB (< 320 KB WGP LDS)

DEVI v8f wmma_bf16(v16bf a, v16bf b, v8f c) {
  return __builtin_amdgcn_wmma_f32_16x16x32_bf16(false, a, false, b, (short)0, c, false, false);
}

// A-matrix fragment (16xK tile, bf16): lane holds row = lane&15;
// element j maps to col K = kbase + (j&7) + 8*half + (j>=8 ? 16 : 0)
// => two contiguous 16B LDS loads.
DEVI v16bf load_a(const __bf16* base, int row, int stride, int kbase, int hf) {
  union { v16bf v; v8bf h[2]; } u;
  const __bf16* p = base + row * stride + kbase + 8 * hf;
  u.h[0] = *(const v8bf*)(p);
  u.h[1] = *(const v8bf*)(p + 16);
  return u.v;
}

// B-matrix fragment (Kx16 tile, bf16) from an N-major (transposed) weight:
// lane holds col N; element j maps to row K = kbase + (j&7) + 8*half + (j>=8 ? 16 : 0)
// => two contiguous 16B LDS loads.
DEVI v16bf load_b(const __bf16* wT, int n, int stride, int kbase, int hf) {
  union { v16bf v; v8bf h[2]; } u;
  const __bf16* p = wT + n * stride + kbase + 8 * hf;
  u.h[0] = *(const v8bf*)(p);
  u.h[1] = *(const v8bf*)(p + 16);
  return u.v;
}

__global__ __launch_bounds__(512, 1)
void dynfilter_kernel(const float* __restrict__ xyz,   const float* __restrict__ xyz_nn,
                      const float* __restrict__ points, const int* __restrict__ knn,
                      const float* __restrict__ dists, const int* __restrict__ mask,
                      const float* __restrict__ W1,  const float* __restrict__ b1,
                      const float* __restrict__ W2,  const float* __restrict__ b2,
                      const float* __restrict__ Wm1, const float* __restrict__ bm1,
                      const float* __restrict__ Wm2, const float* __restrict__ bm2,
                      const float* __restrict__ Ws,  const float* __restrict__ bs,
                      float* __restrict__ out)
{
  extern __shared__ char smem[];
  __bf16* w1T   = (__bf16*)(smem + O_W1T);
  __bf16* w2T   = (__bf16*)(smem + O_W2T);
  __bf16* wm1T  = (__bf16*)(smem + O_WM1T);
  __bf16* wm2T  = (__bf16*)(smem + O_WM2T);
  __bf16* wsT   = (__bf16*)(smem + O_WST);
  __bf16* gp    = (__bf16*)(smem + O_GP);
  __bf16* featA = (__bf16*)(smem + O_FEAT);
  __bf16* pts   = (__bf16*)(smem + O_PTS);
  __bf16* hbuf  = (__bf16*)(smem + O_HB);
  __bf16* hbuf2 = (__bf16*)(smem + O_HB2);
  float*  maskf = (float*)(smem + O_MSK);
  float*  b1f   = (float*)(smem + O_B1);
  float*  b2f   = (float*)(smem + O_B2);
  float*  bm1f  = (float*)(smem + O_BM1);
  float*  bm2f  = (float*)(smem + O_BM2);
  float*  bsf   = (float*)(smem + O_BS);

  const int tid  = threadIdx.x;
  const int wid  = tid >> 5;       // wave id 0..15 == point-in-tile
  const int lane = tid & 31;
  const int hf   = lane >> 4;      // half-wave select
  const int ln   = lane & 15;

  // ---- cooperative weight staging: f32 -> bf16, transposed (N-major), zero-padded ----
  for (int i = tid; i < 32*96; i += 512) { int n = i/96, kk = i%96; w1T[i]  = (__bf16)(kk < 68 ? W1[kk*32 + n]  : 0.f); }
  for (int i = tid; i < 80*32; i += 512) { int n = i/32, kk = i%32; w2T[i]  = (__bf16)(n  < 68 ? W2[kk*68 + n]  : 0.f); }
  for (int i = tid; i < 64*96; i += 512) { int n = i/96, kk = i%96; wm1T[i] = (__bf16)(kk < 68 ? Wm1[kk*64 + n] : 0.f); }
  for (int i = tid; i < 64*64; i += 512) { int n = i/64, kk = i%64; wm2T[i] = (__bf16)Wm2[kk*64 + n];
                                           wsT[i]  = (__bf16)Ws[kk*64 + n]; }
  for (int i = tid; i < 16*96; i += 512) featA[i] = (__bf16)0.f;   // pad cols stay zero
  if (tid < 32) b1f[tid] = b1[tid];
  if (tid < 80) b2f[tid] = (tid < 68) ? b2[tid] : 0.f;
  if (tid < 64) { bm1f[tid] = bm1[tid]; bm2f[tid] = bm2[tid]; bsf[tid] = bs[tid]; }

  // ---- gather: build gp[wave][16 neighbors][96] (bf16) + mask + own-point features ----
  const int  p     = blockIdx.x * 16 + wid;       // flat point id
  const int  bidx  = p >> NSHIFT;
  const long nbase = (long)bidx << NSHIFT;
  {
    const int  k   = ln;
    const long e   = (long)p * 16 + k;
    const int  idx = knn[e];
    const float* prow = points + (nbase + idx) * 64;
    __bf16* g = gp + (wid * 16 + k) * 96;
    if (hf == 0) {
      const float* a3 = xyz_nn + (nbase + idx) * 3;
      const float* x3 = xyz + (long)p * 3;
      g[0] = (__bf16)(a3[0] - x3[0]);
      g[1] = (__bf16)(a3[1] - x3[1]);
      g[2] = (__bf16)(a3[2] - x3[2]);
      g[3] = (__bf16)dists[e];
      maskf[wid * 16 + k] = (float)mask[e];
      #pragma unroll
      for (int c = 0; c < 32; c += 4) {
        float4 v = *(const float4*)(prow + c);
        g[4+c] = (__bf16)v.x; g[5+c] = (__bf16)v.y; g[6+c] = (__bf16)v.z; g[7+c] = (__bf16)v.w;
      }
    } else {
      #pragma unroll
      for (int c = 32; c < 64; c += 4) {
        float4 v = *(const float4*)(prow + c);
        g[4+c] = (__bf16)v.x; g[5+c] = (__bf16)v.y; g[6+c] = (__bf16)v.z; g[7+c] = (__bf16)v.w;
      }
      #pragma unroll
      for (int c = 68; c < 96; c++) g[c] = (__bf16)0.f;   // K-dim zero padding
    }
    const float* my = points + (long)p * 64;            // shortcut input tile
    pts[wid * 64 + lane]      = (__bf16)my[lane];
    pts[wid * 64 + lane + 32] = (__bf16)my[lane + 32];
  }
  __syncthreads();

  // ---- stage 1, GEMM1: H[16x32] = relu(gp[16x96] @ W1[96x32] + b1) ----
  const __bf16* gpw = gp + wid * 16 * 96;
  __bf16* hb = hbuf + wid * 16 * 32;
  #pragma unroll
  for (int t = 0; t < 2; t++) {
    v8f acc = {};
    #pragma unroll
    for (int q = 0; q < 3; q++) {
      v16bf a  = load_a(gpw, ln, 96, 32 * q, hf);
      v16bf bb = load_b(w1T, 16 * t + ln, 96, 32 * q, hf);
      acc = wmma_bf16(a, bb, acc);
    }
    const float bias = b1f[16 * t + ln];
    #pragma unroll
    for (int v = 0; v < 8; v++) {
      float h = acc[v] + bias;
      hb[(v + 8 * hf) * 32 + 16 * t + ln] = (__bf16)(h > 0.f ? h : 0.f);
    }
  }
  __syncthreads();

  // ---- stage 1, GEMM2 + masked softmax over K + weighted sum -> feat[68] ----
  const v16bf ah = load_a(hb, ln, 32, 0, hf);   // H as A-matrix (contraction = 32)
  #pragma unroll
  for (int t = 0; t < 5; t++) {
    v16bf bb = load_b(w2T, 16 * t + ln, 32, 0, hf);
    v8f acc = {};
    acc = wmma_bf16(ah, bb, acc);
    const int   c    = 16 * t + ln;             // channel (column)
    const float bias = b2f[c];
    float dk[8], mx = -3.4e38f;
    #pragma unroll
    for (int v = 0; v < 8; v++) { dk[v] = acc[v] + bias; mx = fmaxf(mx, dk[v]); }
    mx = fmaxf(mx, __shfl_xor(mx, 16, 32));     // combine the two row-halves (same column)
    float e[8], s = 0.f;
    #pragma unroll
    for (int v = 0; v < 8; v++) { e[v] = __expf(dk[v] - mx) * maskf[wid * 16 + v + 8 * hf]; s += e[v]; }
    s += __shfl_xor(s, 16, 32);
    const float inv = 1.f / (s + 1e-8f);
    float fc = 0.f;
    #pragma unroll
    for (int v = 0; v < 8; v++) fc += e[v] * inv * (float)gpw[(v + 8 * hf) * 96 + c];
    fc += __shfl_xor(fc, 16, 32);
    if (hf == 0 && c < 68) featA[wid * 96 + c] = (__bf16)fc;  // row = point-in-tile
  }
  __syncthreads();

  // ---- stage 2A: H2[16x64] = relu(feat[16x96] @ Wm1[96x64] + bm1) (4 waves, one n-tile each) ----
  if (wid < 4) {
    const int t = wid;
    v8f acc = {};
    #pragma unroll
    for (int q = 0; q < 3; q++) {
      v16bf a  = load_a(featA, ln, 96, 32 * q, hf);
      v16bf bb = load_b(wm1T, 16 * t + ln, 96, 32 * q, hf);
      acc = wmma_bf16(a, bb, acc);
    }
    const float bias = bm1f[16 * t + ln];
    #pragma unroll
    for (int v = 0; v < 8; v++) {
      float h = acc[v] + bias;
      hbuf2[(v + 8 * hf) * 64 + 16 * t + ln] = (__bf16)(h > 0.f ? h : 0.f);
    }
  }
  __syncthreads();

  // ---- stage 2B: out = H2 @ Wm2 + bm2 + pts @ Ws + bs ----
  if (wid < 4) {
    const int t = wid;
    v8f acc = {}, acc2 = {};
    #pragma unroll
    for (int q = 0; q < 2; q++) {
      v16bf a  = load_a(hbuf2, ln, 64, 32 * q, hf);
      v16bf bb = load_b(wm2T, 16 * t + ln, 64, 32 * q, hf);
      acc = wmma_bf16(a, bb, acc);
      v16bf a2 = load_a(pts, ln, 64, 32 * q, hf);
      v16bf b2v = load_b(wsT, 16 * t + ln, 64, 32 * q, hf);
      acc2 = wmma_bf16(a2, b2v, acc2);
    }
    const int   col  = 16 * t + ln;
    const float bias = bm2f[col] + bsf[col];
    const long  base = (long)blockIdx.x * 16;
    #pragma unroll
    for (int v = 0; v < 8; v++) {
      const int m = v + 8 * hf;                 // point-in-tile
      out[(base + m) * 64 + col] = acc[v] + acc2[v] + bias;
    }
  }
}

extern "C" void kernel_launch(void* const* d_in, const int* in_sizes, int n_in,
                              void* d_out, int out_size, void* d_ws, size_t ws_size,
                              hipStream_t stream) {
  (void)in_sizes; (void)n_in; (void)out_size; (void)d_ws; (void)ws_size;
  const float* xyz    = (const float*)d_in[0];
  const float* xyz_nn = (const float*)d_in[1];
  const float* points = (const float*)d_in[2];
  const int*   knn    = (const int*)  d_in[3];
  const float* dists  = (const float*)d_in[4];
  const int*   mask   = (const int*)  d_in[5];
  const float* W1  = (const float*)d_in[6];   const float* b1  = (const float*)d_in[7];
  const float* W2  = (const float*)d_in[8];   const float* b2  = (const float*)d_in[9];
  const float* Wm1 = (const float*)d_in[10];  const float* bm1 = (const float*)d_in[11];
  const float* Wm2 = (const float*)d_in[12];  const float* bm2 = (const float*)d_in[13];
  const float* Ws  = (const float*)d_in[14];  const float* bs  = (const float*)d_in[15];
  float* out = (float*)d_out;

  // Allow >64KB dynamic LDS (CDNA5 WGP supports up to 320KB per workgroup).
  static_assert(SMEM_BYTES < 320 * 1024, "LDS overflow");
  hipFuncSetAttribute((const void*)dynfilter_kernel,
                      hipFuncAttributeMaxDynamicSharedMemorySize, SMEM_BYTES);

  const int blocks = BN / 16;   // 8192 tiles of 16 points
  dynfilter_kernel<<<blocks, 512, SMEM_BYTES, stream>>>(
      xyz, xyz_nn, points, knn, dists, mask,
      W1, b1, W2, b2, Wm1, bm1, Wm2, bm2, Ws, bs, out);
}
